// DistributedDotGAT_42064909697462
// MI455X (gfx1250) — matrix operand order repaired and checked by reading, not compiled, not run on hardware
//
#include <hip/hip_runtime.h>
#include <cstdint>
#include <cstddef>

// ---------------- problem constants ----------------
#define B_    4
#define A_    1024
#define DIN_  256
#define DH_   512
#define DOUT_ 128
#define NH_   8
#define STEPS_ 3
#define TOPK_ 5
#define BA_   (B_*A_)

// ---------------- bf16 / WMMA types ----------------
typedef __bf16 bf16_t;
typedef __attribute__((ext_vector_type(16))) __bf16 v16bf;
typedef __attribute__((ext_vector_type(8)))  __bf16 v8bf;
typedef __attribute__((ext_vector_type(8)))  float  v8f;
typedef __attribute__((ext_vector_type(4)))  unsigned int v4u;
typedef __attribute__((ext_vector_type(8)))  int v8i;
typedef __attribute__((ext_vector_type(4)))  int v4i;

union BFrag { v16bf v; v8bf h[2]; };

__device__ inline bf16_t f2bf(float f) {
  union { float f; uint32_t u; } in; in.f = f;
  uint32_t u = in.u;
  uint32_t r = (u + 0x7FFFu + ((u >> 16) & 1u)) >> 16;  // round-nearest-even
  union { uint16_t s; bf16_t b; } out; out.s = (uint16_t)r;
  return out.b;
}
__device__ inline float bf2f(bf16_t x) {
  union { bf16_t b; uint16_t s; } in; in.b = x;
  union { uint32_t u; float f; } out; out.u = ((uint32_t)in.s) << 16;
  return out.f;
}

// ---------------------------------------------------------------------------
// Tensor Data Mover: 2D tile DMA (global -> LDS).
// Tile: tile_dim0 = 32 contiguous bf16 along K, tile_dim1 = 128 rows,
// row stride = ld_elems. D# packed per CDNA5 ISA 8.3/8.4 (count=1, type=2,
// data_size=2B). Groups 2/3 are zero (<=2D tensor). Tracked by TENSORcnt.
// ---------------------------------------------------------------------------
__device__ inline void tdm_load_2d(const bf16_t* gsrc, unsigned lds_addr,
                                   int ld_elems) {
  const unsigned long long ga = (unsigned long long)(uintptr_t)gsrc;
  v4u g0;
  g0[0] = 1u;                                                 // count=1 (valid user D#)
  g0[1] = lds_addr;                                           // lds_addr (bytes)
  g0[2] = (unsigned)ga;                                       // global_addr[31:0]
  g0[3] = (unsigned)((ga >> 32) & 0x01FFFFFFull) | (2u << 30);// global_addr[56:32] | type=2

  const unsigned td0 = (unsigned)ld_elems;                    // tensor_dim0
  const unsigned td1 = 128u;                                  // tensor_dim1 (rows)
  const unsigned long long st0 = (unsigned long long)ld_elems;// tensor_dim0_stride
  v8i g1;
  g1[0] = (int)(1u << 16);                                    // wg_mask=0, data_size=1 (2B)
  g1[1] = (int)((td0 & 0xFFFFu) << 16);                       // [63:48] = tensor_dim0 lo16
  g1[2] = (int)(((td0 >> 16) & 0xFFFFu) | ((td1 & 0xFFFFu) << 16));
  g1[3] = (int)(((td1 >> 16) & 0xFFFFu) | (32u << 16));       // [127:112] tile_dim0 = 32
  g1[4] = (int)128u;                                          // [143:128] tile_dim1 = 128
  g1[5] = (int)(unsigned)st0;                                 // stride0 lo32
  g1[6] = (int)(unsigned)((st0 >> 32) & 0xFFFFull);           // stride0 hi16
  g1[7] = 0;

  v4i gz = {0, 0, 0, 0};
#if __clang_major__ >= 23
  v8i gz8 = {0, 0, 0, 0, 0, 0, 0, 0};
  __builtin_amdgcn_tensor_load_to_lds(g0, g1, gz, gz, gz8, 0);
#else
  __builtin_amdgcn_tensor_load_to_lds(g0, g1, gz, gz, 0);
#endif
}

// ---------------------------------------------------------------------------
// Wave-level NT GEMM core (pure global-load version, used by small GEMMs):
// C[32x64] += A[m0:m0+32, :K] * Bt[n0:n0+64, :K]^T
// Fragment layouts: contiguous k order (k = 16*(lane/16) + s), same for A and B.
// ---------------------------------------------------------------------------
__device__ inline void wave_gemm(const bf16_t* __restrict__ Ap, int lda,
                                 const bf16_t* __restrict__ Btp, int ldb,
                                 int K, int m0, int n0, v8f acc[2][4]) {
  const int lane = threadIdx.x & 31;
  const int lrow = lane & 15;
  const int lhi  = lane >> 4;
  for (int k = 0; k < K; k += 32) {
    if (k + 32 < K) {  // gfx1250 global_prefetch_b8 of next K-slab
      __builtin_prefetch((const void*)(Ap  + (size_t)(m0 + lrow) * lda + (k + 32)), 0, 1);
      __builtin_prefetch((const void*)(Btp + (size_t)(n0 + lrow) * ldb + (k + 32)), 0, 1);
    }
    v16bf af[2];
#pragma unroll
    for (int i = 0; i < 2; ++i) {
      const bf16_t* p = Ap + (size_t)(m0 + i * 16 + lrow) * lda + k + lhi * 16;
      BFrag f;
      f.h[0] = *(const v8bf*)p;
      f.h[1] = *(const v8bf*)(p + 8);
      af[i] = f.v;
    }
    v16bf bfr[4];
#pragma unroll
    for (int j = 0; j < 4; ++j) {
      const bf16_t* p = Btp + (size_t)(n0 + j * 16 + lrow) * ldb + k + lhi * 16;
      BFrag f;
      f.h[0] = *(const v8bf*)p;
      f.h[1] = *(const v8bf*)(p + 8);
      bfr[j] = f.v;
    }
#pragma unroll
    for (int i = 0; i < 2; ++i)
#pragma unroll
      for (int j = 0; j < 4; ++j)
        acc[i][j] = __builtin_amdgcn_wmma_f32_16x16x32_bf16(
            false, af[i], false, bfr[j], (short)0, acc[i][j], false, false);
  }
}

// ---------------------------------------------------------------------------
// TDM-staged NT GEMM core: the block's 128-row B-slab for each 32-wide K-step
// is DMA'd into LDS once (wave 0 issues tensor_load_to_lds, double-buffered),
// all 8 waves read their B fragments from LDS (ds_load_b128). A stays on the
// global path (shared across blocks through L2) with prefetch.
// ---------------------------------------------------------------------------
__device__ inline void wave_gemm_ldsB(const bf16_t* __restrict__ Ap, int lda,
                                      const bf16_t* __restrict__ Btp, int ldb,
                                      int K, int m0, int nblk, v8f acc[2][4]) {
  __shared__ bf16_t ldsB[2][128 * 32];
  const int tid  = threadIdx.x;
  const int wave = tid >> 5;
  const int wn   = wave & 1;
  const int lane = tid & 31;
  const int lrow = lane & 15, lhi = lane >> 4;

  // prologue: stage slab 0
  if (wave == 0) {
    tdm_load_2d(Btp + (size_t)nblk * ldb, (unsigned)(uintptr_t)&ldsB[0][0], ldb);
    __builtin_amdgcn_s_wait_tensorcnt(0);
  }
  __syncthreads();

  for (int k = 0; k < K; k += 32) {
    const int cur = (k >> 5) & 1;
    // kick DMA for the next slab into the other buffer
    if (wave == 0 && (k + 32) < K)
      tdm_load_2d(Btp + (size_t)nblk * ldb + (k + 32),
                  (unsigned)(uintptr_t)&ldsB[cur ^ 1][0], ldb);

    if (k + 32 < K)
      __builtin_prefetch((const void*)(Ap + (size_t)(m0 + lrow) * lda + (k + 32)), 0, 1);

    v16bf af[2];
#pragma unroll
    for (int i = 0; i < 2; ++i) {
      const bf16_t* p = Ap + (size_t)(m0 + i * 16 + lrow) * lda + k + lhi * 16;
      BFrag f;
      f.h[0] = *(const v8bf*)p;
      f.h[1] = *(const v8bf*)(p + 8);
      af[i] = f.v;
    }
    v16bf bfr[4];
#pragma unroll
    for (int j = 0; j < 4; ++j) {
      const bf16_t* p = &ldsB[cur][(wn * 64 + j * 16 + lrow) * 32 + lhi * 16];
      BFrag f;
      f.h[0] = *(const v8bf*)p;      // ds_load_b128
      f.h[1] = *(const v8bf*)(p + 8);
      bfr[j] = f.v;
    }
#pragma unroll
    for (int i = 0; i < 2; ++i)
#pragma unroll
      for (int j = 0; j < 4; ++j)
        acc[i][j] = __builtin_amdgcn_wmma_f32_16x16x32_bf16(
            false, af[i], false, bfr[j], (short)0, acc[i][j], false, false);

    if (wave == 0 && (k + 32) < K) __builtin_amdgcn_s_wait_tensorcnt(0);
    __syncthreads();
  }
}

// 8 waves / block arranged 4x2 -> block tile 128 rows x 128 cols.
__device__ inline void tile_coords(int& m0, int& n0) {
  const int wave = threadIdx.x >> 5;
  m0 = blockIdx.x * 128 + (wave >> 1) * 32;
  n0 = blockIdx.y * 128 + (wave & 1) * 64;
}

// ---------------- GEMM kernels (different epilogues) ----------------

// h = x @ Wi + bi + emb  -> bf16
__global__ __launch_bounds__(256)
void gemm_hinit(const bf16_t* __restrict__ Xbf, const bf16_t* __restrict__ Wit,
                const float* __restrict__ bi, const float* __restrict__ emb,
                bf16_t* __restrict__ Hbf) {
  v8f acc[2][4] = {};
  int m0, n0; tile_coords(m0, n0);
  wave_gemm(Xbf, DIN_, Wit, DIN_, DIN_, m0, n0, acc);
  const int lane = threadIdx.x & 31, lrow = lane & 15, lhi = lane >> 4;
#pragma unroll
  for (int i = 0; i < 2; ++i)
#pragma unroll
    for (int j = 0; j < 4; ++j)
#pragma unroll
      for (int r = 0; r < 8; ++r) {
        const int rowi = m0 + i * 16 + r + 8 * lhi;
        const int col  = n0 + j * 16 + lrow;
        const float v = acc[i][j][r] + bi[col] + emb[(size_t)(rowi & (A_ - 1)) * DH_ + col];
        Hbf[(size_t)rowi * DH_ + col] = f2bf(v);
      }
}

// Q/K/V = h @ Wq/Wk/Wv  (z = which*NH + head) -> bf16    [TDM-staged B]
__global__ __launch_bounds__(256)
void gemm_qkv(const bf16_t* __restrict__ Hbf, const bf16_t* __restrict__ Wt,
              bf16_t* __restrict__ QKV) {
  const int z = blockIdx.z;
  const bf16_t* W = Wt + (size_t)z * DH_ * DH_;
  bf16_t* Out = QKV + (size_t)z * BA_ * DH_;
  v8f acc[2][4] = {};
  const int wave = threadIdx.x >> 5;
  const int m0   = blockIdx.x * 128 + (wave >> 1) * 32;
  const int nblk = blockIdx.y * 128;
  wave_gemm_ldsB(Hbf, DH_, W, DH_, DH_, m0, nblk, acc);
  const int n0 = nblk + (wave & 1) * 64;
  const int lane = threadIdx.x & 31, lrow = lane & 15, lhi = lane >> 4;
#pragma unroll
  for (int i = 0; i < 2; ++i)
#pragma unroll
    for (int j = 0; j < 4; ++j)
#pragma unroll
      for (int r = 0; r < 8; ++r) {
        const int rowi = m0 + i * 16 + r + 8 * lhi;
        const int col  = n0 + j * 16 + lrow;
        Out[(size_t)rowi * DH_ + col] = f2bf(acc[i][j][r]);
      }
}

// S[h,b] = Q[h,b] @ K[h,b]^T / sqrt(DH) + conn   (z = head*B + b)  [TDM-staged K]
__global__ __launch_bounds__(256)
void gemm_scores(const bf16_t* __restrict__ QKV, const float* __restrict__ conn,
                 float* __restrict__ Sc) {
  const int z = blockIdx.z;
  const int head = z / B_, b = z % B_;
  const bf16_t* Qp = QKV + ((size_t)head * BA_ + (size_t)b * A_) * DH_;
  const bf16_t* Kp = QKV + (((size_t)(NH_ + head)) * BA_ + (size_t)b * A_) * DH_;
  float* S = Sc + (size_t)z * A_ * A_;
  v8f acc[2][4] = {};
  const int wave = threadIdx.x >> 5;
  const int m0   = blockIdx.x * 128 + (wave >> 1) * 32;
  const int nblk = blockIdx.y * 128;
  wave_gemm_ldsB(Qp, DH_, Kp, DH_, DH_, m0, nblk, acc);
  const int n0 = nblk + (wave & 1) * 64;
  const float inv_scale = 0.04419417382415922f;  // 1/sqrt(512)
  const int lane = threadIdx.x & 31, lrow = lane & 15, lhi = lane >> 4;
#pragma unroll
  for (int i = 0; i < 2; ++i)
#pragma unroll
    for (int j = 0; j < 4; ++j)
#pragma unroll
      for (int r = 0; r < 8; ++r) {
        const int rowi = m0 + i * 16 + r + 8 * lhi;
        const int col  = n0 + j * 16 + lrow;
        S[(size_t)rowi * A_ + col] = acc[i][j][r] * inv_scale + conn[(size_t)rowi * A_ + col];
      }
}

// out = h @ Wo + bo -> f32
__global__ __launch_bounds__(256)
void gemm_final(const bf16_t* __restrict__ Hbf, const bf16_t* __restrict__ Wot,
                const float* __restrict__ bo, float* __restrict__ Out) {
  v8f acc[2][4] = {};
  int m0, n0; tile_coords(m0, n0);
  wave_gemm(Hbf, DH_, Wot, DH_, DH_, m0, n0, acc);
  const int lane = threadIdx.x & 31, lrow = lane & 15, lhi = lane >> 4;
#pragma unroll
  for (int i = 0; i < 2; ++i)
#pragma unroll
    for (int j = 0; j < 4; ++j)
#pragma unroll
      for (int r = 0; r < 8; ++r) {
        const int rowi = m0 + i * 16 + r + 8 * lhi;
        const int col  = n0 + j * 16 + lrow;
        Out[(size_t)rowi * DOUT_ + col] = acc[i][j][r] + bo[col];
      }
}

// ---------------------------------------------------------------------------
// Per (head,batch,row): top-5 over 1024 scores, softmax over 5, sparse gather
// of 5 V rows, LayerNorm, atomic accumulate of head-mean into Hacc.
// ---------------------------------------------------------------------------
__device__ inline void ins5(float v, int ix, float* tv, int* ti) {
  if (v > tv[TOPK_ - 1]) {
    tv[TOPK_ - 1] = v; ti[TOPK_ - 1] = ix;
#pragma unroll
    for (int j = TOPK_ - 1; j > 0; --j)
      if (tv[j] > tv[j - 1]) {
        float tf = tv[j]; tv[j] = tv[j - 1]; tv[j - 1] = tf;
        int td = ti[j]; ti[j] = ti[j - 1]; ti[j - 1] = td;
      }
  }
}

__global__ __launch_bounds__(256)
void attn_topk_ln(const float* __restrict__ Sc, const bf16_t* __restrict__ QKV,
                  const float* __restrict__ gamma, const float* __restrict__ beta,
                  float* __restrict__ Hacc) {
  const int tid = threadIdx.x;
  const int a = blockIdx.x;
  const int z = blockIdx.y;            // head*B + b
  const int head = z / B_, b = z % B_;
  const float* row = Sc + ((size_t)z * A_ + a) * A_;

  __shared__ float svals[256 * TOPK_];
  __shared__ int   sidx [256 * TOPK_];
  __shared__ float mvals[32 * TOPK_];
  __shared__ int   midx [32 * TOPK_];
  __shared__ float wts[TOPK_];
  __shared__ int   wid[TOPK_];
  __shared__ float red[256];
  __shared__ float stat[2];

  // phase 1: per-thread top-5 over strided columns
  float tv[TOPK_]; int ti[TOPK_];
#pragma unroll
  for (int j = 0; j < TOPK_; ++j) { tv[j] = -3.0e38f; ti[j] = 0; }
  for (int c = tid; c < A_; c += 256) ins5(row[c], c, tv, ti);
#pragma unroll
  for (int j = 0; j < TOPK_; ++j) { svals[tid * TOPK_ + j] = tv[j]; sidx[tid * TOPK_ + j] = ti[j]; }
  __syncthreads();

  // phase 2: 32 threads merge 8 lists each
  if (tid < 32) {
    float mv[TOPK_]; int mi[TOPK_];
#pragma unroll
    for (int j = 0; j < TOPK_; ++j) { mv[j] = -3.0e38f; mi[j] = 0; }
    for (int l = 0; l < 8; ++l) {
      const int base = (tid * 8 + l) * TOPK_;
      for (int j = 0; j < TOPK_; ++j) ins5(svals[base + j], sidx[base + j], mv, mi);
    }
    for (int j = 0; j < TOPK_; ++j) { mvals[tid * TOPK_ + j] = mv[j]; midx[tid * TOPK_ + j] = mi[j]; }
  }
  __syncthreads();

  // phase 3: final merge + softmax over 5
  if (tid == 0) {
    float fv[TOPK_]; int fi[TOPK_];
#pragma unroll
    for (int j = 0; j < TOPK_; ++j) { fv[j] = -3.0e38f; fi[j] = 0; }
    for (int l = 0; l < 32; ++l)
      for (int j = 0; j < TOPK_; ++j) ins5(mvals[l * TOPK_ + j], midx[l * TOPK_ + j], fv, fi);
    const float m = fv[0];
    float e[TOPK_]; float s = 0.f;
#pragma unroll
    for (int j = 0; j < TOPK_; ++j) { e[j] = expf(fv[j] - m); s += e[j]; }
    const float inv = 1.f / s;
#pragma unroll
    for (int j = 0; j < TOPK_; ++j) { wts[j] = e[j] * inv; wid[j] = fi[j]; }
  }
  __syncthreads();

  // phase 4: sparse alpha @ V (5 rows), then LayerNorm across DH
  const bf16_t* Vp = QKV + (((size_t)(2 * NH_ + head)) * BA_ + (size_t)b * A_) * DH_;
  float o[2];
#pragma unroll
  for (int e = 0; e < 2; ++e) {
    const int d = e * 256 + tid;
    float s = 0.f;
#pragma unroll
    for (int j = 0; j < TOPK_; ++j)
      s += wts[j] * bf2f(Vp[(size_t)wid[j] * DH_ + d]);
    o[e] = s;
  }
  red[tid] = o[0] + o[1];
  __syncthreads();
  for (int s = 128; s > 0; s >>= 1) { if (tid < s) red[tid] += red[tid + s]; __syncthreads(); }
  if (tid == 0) stat[0] = red[0] * (1.0f / DH_);
  __syncthreads();
  red[tid] = o[0] * o[0] + o[1] * o[1];
  __syncthreads();
  for (int s = 128; s > 0; s >>= 1) { if (tid < s) red[tid] += red[tid + s]; __syncthreads(); }
  if (tid == 0) {
    const float mu = stat[0];
    const float var = red[0] * (1.0f / DH_) - mu * mu;
    stat[1] = rsqrtf(var + 1e-5f);
  }
  __syncthreads();
  const float mu = stat[0], rstd = stat[1];
#pragma unroll
  for (int e = 0; e < 2; ++e) {
    const int d = e * 256 + tid;
    const float val = (o[e] - mu) * rstd * gamma[head * DH_ + d] + beta[head * DH_ + d];
    atomicAdd(&Hacc[((size_t)b * A_ + a) * DH_ + d], val * (1.0f / NH_));
  }
}

// ---------------- small utility kernels ----------------
__global__ void convert_f32_bf16(const float* __restrict__ in, bf16_t* __restrict__ out, int n) {
  const int i = blockIdx.x * blockDim.x + threadIdx.x;
  if (i < n) out[i] = f2bf(in[i]);
}

__global__ void fill_zero_f32(float* __restrict__ p, int n) {
  const int i = blockIdx.x * blockDim.x + threadIdx.x;
  if (i < n) p[i] = 0.0f;
}

__global__ void swish_to_bf16(const float* __restrict__ in, bf16_t* __restrict__ out, int n) {
  const int i = blockIdx.x * blockDim.x + threadIdx.x;
  if (i < n) {
    const float s = in[i];
    out[i] = f2bf(s * (1.0f / (1.0f + expf(-s))));
  }
}

// transpose-convert: out[n*K + k] = bf16(in[z*K*N + k*N + n]) ; 16x16 LDS tiles
__global__ __launch_bounds__(256)
void transpose_to_bf16(const float* __restrict__ in, bf16_t* __restrict__ out, int K, int N) {
  __shared__ float tile[16][17];
  const float* src = in + (size_t)blockIdx.z * K * N;
  bf16_t* dst = out + (size_t)blockIdx.z * K * N;
  const int tx = threadIdx.x & 15, ty = threadIdx.x >> 4;
  const int k0 = blockIdx.y * 16, n0 = blockIdx.x * 16;
  tile[ty][tx] = src[(size_t)(k0 + ty) * N + (n0 + tx)];
  __syncthreads();
  dst[(size_t)(n0 + ty) * K + (k0 + tx)] = f2bf(tile[tx][ty]);
}

// ---------------------------------------------------------------------------
extern "C" void kernel_launch(void* const* d_in, const int* in_sizes, int n_in,
                              void* d_out, int out_size, void* d_ws, size_t ws_size,
                              hipStream_t stream) {
  (void)in_sizes; (void)n_in; (void)out_size; (void)ws_size;
  const float* x     = (const float*)d_in[0];
  const float* Wi    = (const float*)d_in[1];
  const float* bi    = (const float*)d_in[2];
  const float* emb   = (const float*)d_in[3];
  const float* conn  = (const float*)d_in[4];
  const float* Wq    = (const float*)d_in[5];
  const float* Wk    = (const float*)d_in[6];
  const float* Wv    = (const float*)d_in[7];
  const float* gamma = (const float*)d_in[8];
  const float* beta  = (const float*)d_in[9];
  const float* Wo    = (const float*)d_in[10];
  const float* bo    = (const float*)d_in[11];
  float* out = (float*)d_out;

  // workspace carve-out (~262 MB)
  char* ws = (char*)d_ws;
  size_t off = 0;
  auto take = [&](size_t bytes) -> char* {
    char* p = ws + off; off += (bytes + 255) & ~(size_t)255; return p;
  };
  bf16_t* x_bf  = (bf16_t*)take((size_t)BA_ * DIN_ * 2);
  bf16_t* Wit   = (bf16_t*)take((size_t)DH_ * DIN_ * 2);
  bf16_t* Wqkvt = (bf16_t*)take((size_t)3 * NH_ * DH_ * DH_ * 2);
  bf16_t* Wot   = (bf16_t*)take((size_t)DOUT_ * DH_ * 2);
  bf16_t* Hbf   = (bf16_t*)take((size_t)BA_ * DH_ * 2);
  float*  Hacc  = (float*) take((size_t)BA_ * DH_ * 4);
  bf16_t* QKVb  = (bf16_t*)take((size_t)3 * NH_ * BA_ * DH_ * 2);
  float*  Sc    = (float*) take((size_t)NH_ * B_ * A_ * A_ * 4);

  // one-time precision conversion / weight transposition
  convert_f32_bf16<<<(BA_ * DIN_ + 255) / 256, 256, 0, stream>>>(x, x_bf, BA_ * DIN_);
  transpose_to_bf16<<<dim3(DH_ / 16, DIN_ / 16, 1), 256, 0, stream>>>(Wi, Wit, DIN_, DH_);
  transpose_to_bf16<<<dim3(DH_ / 16, DH_ / 16, NH_), 256, 0, stream>>>(Wq, Wqkvt + (size_t)0 * NH_ * DH_ * DH_, DH_, DH_);
  transpose_to_bf16<<<dim3(DH_ / 16, DH_ / 16, NH_), 256, 0, stream>>>(Wk, Wqkvt + (size_t)1 * NH_ * DH_ * DH_, DH_, DH_);
  transpose_to_bf16<<<dim3(DH_ / 16, DH_ / 16, NH_), 256, 0, stream>>>(Wv, Wqkvt + (size_t)2 * NH_ * DH_ * DH_, DH_, DH_);
  transpose_to_bf16<<<dim3(DOUT_ / 16, DH_ / 16, 1), 256, 0, stream>>>(Wo, Wot, DH_, DOUT_);

  // h = x @ Wi + bi + emb
  gemm_hinit<<<dim3(BA_ / 128, DH_ / 128, 1), 256, 0, stream>>>(x_bf, Wit, bi, emb, Hbf);

  for (int step = 0; step < STEPS_; ++step) {
    gemm_qkv<<<dim3(BA_ / 128, DH_ / 128, 3 * NH_), 256, 0, stream>>>(Hbf, Wqkvt, QKVb);
    gemm_scores<<<dim3(A_ / 128, A_ / 128, NH_ * B_), 256, 0, stream>>>(QKVb, conn, Sc);
    fill_zero_f32<<<(BA_ * DH_ + 255) / 256, 256, 0, stream>>>(Hacc, BA_ * DH_);
    attn_topk_ln<<<dim3(A_, NH_ * B_), 256, 0, stream>>>(Sc, QKVb, gamma, beta, Hacc);
    swish_to_bf16<<<(BA_ * DH_ + 255) / 256, 256, 0, stream>>>(Hacc, Hbf, BA_ * DH_);
  }

  // out = h @ Wo + bo
  gemm_final<<<dim3(BA_ / 128, DOUT_ / 128, 1), 256, 0, stream>>>(Hbf, Wot, bo, out);
}
